// Model_37340445671805
// MI455X (gfx1250) — matrix operand order repaired and checked by reading, not compiled, not run on hardware
//
#include <hip/hip_runtime.h>
#include <hip/hip_bf16.h>
#include <cstdint>
#include <cstddef>

// ---------------------------------------------------------------------------
// Types for CDNA5 WMMA (wave32): 16x16x32 bf16 -> f32, and TDM descriptors
// ---------------------------------------------------------------------------
typedef __attribute__((ext_vector_type(16))) __bf16 bf16x16;
typedef __attribute__((ext_vector_type(8)))  float  f32x8;
typedef __attribute__((ext_vector_type(4)))  unsigned int v4u;
typedef __attribute__((ext_vector_type(4)))  int    v4i;
typedef __attribute__((ext_vector_type(8)))  int    v8i;

static inline int cdiv(int a, int b) { return (a + b - 1) / b; }

// Native f32 -> bf16 (lowers to v_cvt_pk_bf16_f32 on gfx1250)
static __device__ __forceinline__ __bf16 f2bf(float x) { return (__bf16)x; }

static __device__ __forceinline__ void pack8(bf16x16& d, int base, float4 a, float4 b) {
  d[base + 0] = f2bf(a.x); d[base + 1] = f2bf(a.y);
  d[base + 2] = f2bf(a.z); d[base + 3] = f2bf(a.w);
  d[base + 4] = f2bf(b.x); d[base + 5] = f2bf(b.y);
  d[base + 6] = f2bf(b.z); d[base + 7] = f2bf(b.w);
}

// ---------------------------------------------------------------------------
// Generic NT GEMM:  C[m,n] = act( sum_k A[m,k]*B[n,k] + bias[n] ) (+ resid[m,n])
// One wave computes a 16x64 tile: 4 accumulators share one A fragment ->
// 4 v_wmma per 32-wide K step.  Main loop uses b128 loads (all lda/ldb used
// by this model are multiples of 4 floats).  act: 0 = none, 1 = softplus.
// ---------------------------------------------------------------------------
__global__ __launch_bounds__(32)
void gemm_nt_wmma(const float* __restrict__ A, int lda,
                  const float* __restrict__ B, int ldb,
                  float* __restrict__ C, int ldc,
                  int M, int N, int K,
                  const float* __restrict__ bias,
                  const float* __restrict__ resid,
                  int act)
{
  const int n0   = blockIdx.x * 64;
  const int m0   = blockIdx.y * 16;
  const int lane = threadIdx.x & 31;
  const int half = lane >> 4;
  const int l15  = lane & 15;

  int arow = m0 + l15; if (arow > M - 1) arow = M - 1;   // clamp: bad rows never stored
  const float* ap = A + (size_t)arow * lda + half * 16;  // A frag K offsets (contiguous: k = 16*half + s)

  const float* bp[4];
#pragma unroll
  for (int j = 0; j < 4; ++j) {
    int c = n0 + j * 16 + l15;
    if (c > N - 1) c = N - 1;                            // clamp: bad cols never stored
    bp[j] = B + (size_t)c * ldb + half * 16;             // B frag K offsets
  }

  const f32x8 vzero = {0.f, 0.f, 0.f, 0.f, 0.f, 0.f, 0.f, 0.f};
  f32x8 acc[4];
#pragma unroll
  for (int j = 0; j < 4; ++j) acc[j] = vzero;

  const int Kmain = K & ~31;
  for (int kb = 0; kb < Kmain; kb += 32) {
    __builtin_prefetch(ap + kb + 256, 0, 0);             // global_prefetch_b8
    const float4* a4 = reinterpret_cast<const float4*>(ap + kb);
    float4 al0 = a4[0], al1 = a4[1], ah0 = a4[2], ah1 = a4[3];
    bf16x16 af;
    pack8(af, 0, al0, al1);
    pack8(af, 8, ah0, ah1);
#pragma unroll
    for (int j = 0; j < 4; ++j) {
      const float4* b4 = reinterpret_cast<const float4*>(bp[j] + kb);
      float4 b0 = b4[0], b1 = b4[1], b2 = b4[2], b3 = b4[3];
      bf16x16 bfv;
      pack8(bfv, 0, b0, b1);
      pack8(bfv, 8, b2, b3);
      acc[j] = __builtin_amdgcn_wmma_f32_16x16x32_bf16(false, af, false, bfv,
                                                       (short)0, acc[j], false, false);
    }
  }
  if (Kmain < K) {  // K tail with zero fill (scalar guarded loads)
    const int kb = Kmain;
    bf16x16 af;
#pragma unroll
    for (int i = 0; i < 8; ++i) {
      int k0 = kb + half * 16 + i;
      int k1 = k0 + 8;
      af[i]     = f2bf(k0 < K ? ap[kb + i]      : 0.f);
      af[i + 8] = f2bf(k1 < K ? ap[kb + 8 + i]  : 0.f);
    }
#pragma unroll
    for (int j = 0; j < 4; ++j) {
      bf16x16 bfv;
#pragma unroll
      for (int i = 0; i < 16; ++i) {
        int k = kb + half * 16 + i;
        bfv[i] = f2bf(k < K ? bp[j][kb + i] : 0.f);
      }
      acc[j] = __builtin_amdgcn_wmma_f32_16x16x32_bf16(false, af, false, bfv,
                                                       (short)0, acc[j], false, false);
    }
  }

#pragma unroll
  for (int j = 0; j < 4; ++j) {
    int n = n0 + j * 16 + l15;
    if (n >= N) continue;
    const float bv = bias ? bias[n] : 0.f;
#pragma unroll
    for (int r = 0; r < 8; ++r) {
      int m = m0 + half * 8 + r;
      if (m < M) {
        float v = acc[j][r] + bv;
        if (act == 1) v = (v > 20.f) ? v : logf(1.f + expf(v));
        if (resid) v += resid[(size_t)m * ldc + n];
        C[(size_t)m * ldc + n] = v;
      }
    }
  }
}

// ---------------------------------------------------------------------------
// Patchify conv as implicit GEMM: out[b,co,p] = sum_{ci,kh,kw} x*w + bias[co]
// K = 128*16*16 = 32768; 16 consecutive K == one 16-wide row of a patch.
// ---------------------------------------------------------------------------
__global__ __launch_bounds__(32)
void patchify_wmma(const float* __restrict__ x, const float* __restrict__ w,
                   const float* __restrict__ bias, float* __restrict__ out)
{
  const int b    = blockIdx.z;
  const int m0   = blockIdx.y * 16;
  const int n0   = blockIdx.x * 16;
  const int lane = threadIdx.x & 31;
  const int half = lane >> 4;
  const int l15  = lane & 15;

  const int co = m0 + l15;                       // 128 % 16 == 0, always valid
  int p = n0 + l15; int pc = p < 196 ? p : 195;  // clamp for loads
  const int ph = pc / 14, pw = pc % 14;

  const float* wrow = w + (size_t)co * 32768 + half * 16;
  const float* xb   = x + (size_t)b * 128 * 224 * 224
                        + (size_t)(ph * 16) * 224 + pw * 16;

  f32x8 acc = {0.f, 0.f, 0.f, 0.f, 0.f, 0.f, 0.f, 0.f};

  for (int kb = 0; kb < 32768; kb += 32) {
    const float4* w4 = reinterpret_cast<const float4*>(wrow + kb);
    float4 wa = w4[0], wb = w4[1], wc = w4[2], wd = w4[3];
    bf16x16 af;
    pack8(af, 0, wa, wb);
    pack8(af, 8, wc, wd);

    const int kk0 = kb + half * 16;
    const int ci  = kk0 >> 8;          // k = ci*256 + kh*16 + kw
    const int kh  = (kk0 >> 4) & 15;
    const float4* x4 = reinterpret_cast<const float4*>(
        xb + (size_t)ci * 224 * 224 + (size_t)kh * 224);
    float4 xa = x4[0], xv = x4[1], xc = x4[2], xd = x4[3];
    bf16x16 bfv;
    pack8(bfv, 0, xa, xv);
    pack8(bfv, 8, xc, xd);

    acc = __builtin_amdgcn_wmma_f32_16x16x32_bf16(false, af, false, bfv,
                                                  (short)0, acc, false, false);
  }

  const int n = n0 + l15;
#pragma unroll
  for (int r = 0; r < 8; ++r) {
    int m = m0 + half * 8 + r;
    if (n < 196) out[((size_t)b * 128 + m) * 196 + n] = acc[r] + bias[m];
  }
}

// ---------------------------------------------------------------------------
// Device->device copy staged through LDS with CDNA5 async loads:
// global_load_async_to_lds_b128 (ASYNCcnt) -> s_wait_asynccnt -> ds_load -> store
// `stage` is the only LDS object, so its base offset is 0: lds addr = tid*16.
// ---------------------------------------------------------------------------
__global__ __launch_bounds__(256)
void async_copy_k(const float4* __restrict__ src, float4* __restrict__ dst, int n4)
{
  __shared__ float4 stage[256];
  int i = blockIdx.x * 256 + threadIdx.x;
  if (i < n4) {
    unsigned lds_addr = threadIdx.x * 16u;
    unsigned long long ga = (unsigned long long)(size_t)(src + i);
    asm volatile("global_load_async_to_lds_b128 %0, %1, off"
                 :: "v"(lds_addr), "v"(ga)
                 : "memory");
    asm volatile("s_wait_asynccnt 0x0" ::: "memory");
    dst[i] = stage[threadIdx.x];
  }
}

// ---------------------------------------------------------------------------
// Device->device copy via the Tensor Data Mover (TENSORcnt path).
// One wave per block; the wave builds a D# (ISA ch.8): group0 holds
// {count=1, lds_addr=0, global_addr, type=2}, group1 holds
// {data_size=4B, 1-row tile of `elems` elements}.  tensor_load_to_lds pulls
// the tile into LDS, s_wait_tensorcnt 0 synchronizes, lanes drain LDS->global.
// ---------------------------------------------------------------------------
#define TDM_TILE 2048
__global__ __launch_bounds__(32)
void tdm_copy_k(const float* __restrict__ src, float* __restrict__ dst, int nfloats)
{
  __shared__ float stage[TDM_TILE];
  const int base = blockIdx.x * TDM_TILE;
  int elems = nfloats - base; if (elems > TDM_TILE) elems = TDM_TILE;

  unsigned long long ga = (unsigned long long)(size_t)(src + base);
  unsigned ue = (unsigned)elems;

  v4u g0;
  g0[0] = 1u;                                            // count=1, user descriptor
  g0[1] = 0u;                                            // lds_addr = 0 (stage base)
  g0[2] = (unsigned)(ga & 0xffffffffu);                  // global_addr[31:0]
  g0[3] = (unsigned)((ga >> 32) & 0x01ffffffu) | (2u << 30);  // addr[56:32] | type=2

  v8i g1;
  g1[0] = (int)(2u << 16);               // wg_mask=0, data_size=2 (4 bytes)
  g1[1] = (int)((ue & 0xffffu) << 16);   // abar=0 | tensor_dim0[15:0]
  g1[2] = (int)((ue >> 16) | (1u << 16));// tensor_dim0[31:16] | tensor_dim1=1
  g1[3] = (int)((ue & 0xffffu) << 16);   // tensor_dim1 hi=0 | tile_dim0
  g1[4] = 0;                             // tile_dim1=0 (unused), tile_dim2=0
  g1[5] = (int)ue;                       // tensor_dim0_stride[31:0]
  g1[6] = 0;                             // stride hi | tensor_dim1_stride lo
  g1[7] = 0;

  v4i gz = {0, 0, 0, 0};

#if __has_builtin(__builtin_amdgcn_tensor_load_to_lds)
#if defined(__clang_major__) && __clang_major__ >= 23
  v8i gz8 = {0, 0, 0, 0, 0, 0, 0, 0};
  __builtin_amdgcn_tensor_load_to_lds(g0, g1, gz, gz, gz8, 0);
#else
  __builtin_amdgcn_tensor_load_to_lds(g0, g1, gz, gz, 0);
#endif
  __builtin_amdgcn_s_wait_tensorcnt(0);
#else
  (void)g0; (void)g1; (void)gz;
  for (int i = threadIdx.x; i < elems; i += 32) stage[i] = src[base + i];
  __builtin_amdgcn_s_barrier();
#endif

  // Launder the LDS pointer so the compiler cannot assume `stage` is unwritten.
  float* sp = stage;
  asm volatile("" : "+v"(sp) :: "memory");
  const float4* s4 = (const float4*)sp;
  float4* d4 = (float4*)(dst + base);
  const int n4 = elems >> 2;
  for (int i = threadIdx.x; i < n4; i += 32) d4[i] = s4[i];
}

// ---------------------------------------------------------------------------
// Fused focus-mix + exact GELU + concat + LayerNorm(768) (LDS block reduce)
// ---------------------------------------------------------------------------
__global__ __launch_bounds__(256)
void build_cc_ln_k(const float* __restrict__ xp, const float* __restrict__ xq,
                   const float* __restrict__ focus, const float* __restrict__ lnw,
                   const float* __restrict__ lnb, float* __restrict__ cc)
{
  const int row = blockIdx.x;      // b*128 + c
  const int t   = threadIdx.x;     // 0..255
  const float f = focus[0];
  float vp = xp[(size_t)row * 256 + t];
  float vq = xq[(size_t)row * 256 + t];
  float xf = f * vq + (2.f - f) * vp;
  xf = 0.5f * xf * (1.f + erff(xf * 0.70710678118f));

  __shared__ float s1[256], s2[256];
  s1[t] = vp + xf + vq;
  s2[t] = vp * vp + xf * xf + vq * vq;
  __syncthreads();
  for (int off = 128; off > 0; off >>= 1) {
    if (t < off) { s1[t] += s1[t + off]; s2[t] += s2[t + off]; }
    __syncthreads();
  }
  const float mu  = s1[0] * (1.f / 768.f);
  const float var = s2[0] * (1.f / 768.f) - mu * mu;
  const float inv = rsqrtf(var + 1e-5f);
  float* o = cc + (size_t)row * 768;
  o[t]       = (vp - mu) * inv * lnw[t]       + lnb[t];
  o[256 + t] = (xf - mu) * inv * lnw[256 + t] + lnb[256 + t];
  o[512 + t] = (vq - mu) * inv * lnw[512 + t] + lnb[512 + t];
}

// ---------------------------------------------------------------------------
// Depthwise causal conv (k=4) along seq + SiLU.  xz rows (b*L) x 2E, take xi.
// ---------------------------------------------------------------------------
__global__ void dwconv_silu_k(const float* __restrict__ xz, int twoE, int E, int L,
                              const float* __restrict__ cw, const float* __restrict__ cb,
                              float* __restrict__ xc, int total)
{
  int idx = blockIdx.x * blockDim.x + threadIdx.x;
  if (idx >= total) return;
  int e = idx % E;
  int l = (idx / E) % L;
  int b = idx / (E * L);
  float s = cb[e];
  const float* base = xz + (size_t)b * L * twoE + e;
#pragma unroll
  for (int j = 0; j < 4; ++j) {
    int ls = l - 3 + j;
    if (ls >= 0) s += base[(size_t)ls * twoE] * cw[e * 4 + j];
  }
  float sig = 1.f / (1.f + expf(-s));
  xc[((size_t)b * L + l) * E + e] = s * sig;
}

// ---------------------------------------------------------------------------
// Selective scan: one thread per (b,e), 16-wide state in registers.
// ---------------------------------------------------------------------------
__global__ void sscan_k(const float* __restrict__ dt, const float* __restrict__ xdbl,
                        int ldx, int dtr,
                        const float* __restrict__ u, const float* __restrict__ Alog,
                        const float* __restrict__ Dp, float* __restrict__ y,
                        int Bn, int L, int E)
{
  int idx = blockIdx.x * blockDim.x + threadIdx.x;
  if (idx >= Bn * E) return;
  int b = idx / E, e = idx % E;
  float A[16], h[16];
#pragma unroll
  for (int s = 0; s < 16; ++s) { A[s] = -expf(Alog[e * 16 + s]); h[s] = 0.f; }
  const float Dv = Dp[e];
  for (int l = 0; l < L; ++l) {
    size_t row = (size_t)b * L + l;
    float dtv = dt[row * E + e];
    float uv  = u[row * E + e];
    float du  = dtv * uv;
    const float* bc = xdbl + row * ldx + dtr;   // Bm at [0..15], Cm at [16..31]
    float accv = 0.f;
#pragma unroll
    for (int s = 0; s < 16; ++s) {
      float dA = expf(dtv * A[s]);
      h[s] = h[s] * dA + du * bc[s];
      accv += h[s] * bc[16 + s];
    }
    y[row * E + e] = accv + uv * Dv;
  }
}

__global__ void gate_silu_k(float* __restrict__ y, const float* __restrict__ xz,
                            int twoE, int E, int total)
{
  int idx = blockIdx.x * blockDim.x + threadIdx.x;
  if (idx >= total) return;
  int e = idx % E;
  size_t r = (size_t)(idx / E);
  float z = xz[r * twoE + E + e];
  y[idx] *= z / (1.f + expf(-z));
}

// ---------------------------------------------------------------------------
// Data-movement helpers
// ---------------------------------------------------------------------------
__global__ void flip_copy_k(const float* __restrict__ src, float* __restrict__ dst,
                            int L, int Dd, int total)
{
  int idx = blockIdx.x * blockDim.x + threadIdx.x;
  if (idx >= total) return;
  int d = idx % Dd;
  int l = (idx / Dd) % L;
  int b = idx / (Dd * L);
  dst[((size_t)b * L + (L - 1 - l)) * Dd + d] = src[idx];
}

__global__ void flip_add_k(float* __restrict__ dst, const float* __restrict__ src,
                           int L, int Dd, int total)
{
  int idx = blockIdx.x * blockDim.x + threadIdx.x;
  if (idx >= total) return;
  int d = idx % Dd;
  int l = (idx / Dd) % L;
  int b = idx / (Dd * L);
  dst[idx] += src[((size_t)b * L + (L - 1 - l)) * Dd + d];
}

__global__ void transpose12_k(const float* __restrict__ src, float* __restrict__ dst,
                              int L, int Dd, int total)
{
  int idx = blockIdx.x * blockDim.x + threadIdx.x;   // src (Bn,L,Dd) -> dst (Bn,Dd,L)
  if (idx >= total) return;
  int d = idx % Dd;
  int l = (idx / Dd) % L;
  int b = idx / (Dd * L);
  dst[((size_t)b * Dd + d) * L + l] = src[idx];
}

// dst (Bn,L,Dd) += src (Bn,Dd,L); optionally flip dst's L axis when indexing src
__global__ void add_trans_k(float* __restrict__ dst, const float* __restrict__ src,
                            int L, int Dd, int flipl, int total)
{
  int idx = blockIdx.x * blockDim.x + threadIdx.x;
  if (idx >= total) return;
  int d = idx % Dd;
  int l = (idx / Dd) % L;
  int b = idx / (Dd * L);
  int ls = flipl ? (L - 1 - l) : l;
  dst[idx] += src[((size_t)b * Dd + d) * L + ls];
}

__global__ void mean_axis1_k(const float* __restrict__ x, float* __restrict__ out,
                             int Bn, int L, int Dd)
{
  int idx = blockIdx.x * blockDim.x + threadIdx.x;
  if (idx >= Bn * Dd) return;
  int b = idx / Dd, d = idx % Dd;
  float s = 0.f;
  for (int l = 0; l < L; ++l) s += x[((size_t)b * L + l) * Dd + d];
  out[idx] = s / (float)L;
}

// ---------------------------------------------------------------------------
// Host-side mamba stack driver (2 layers, residual)
// ---------------------------------------------------------------------------
struct MParams {
  const float *in_w, *conv_w, *conv_b, *xproj_w, *dt_w, *dt_b, *Alog, *D, *out_w;
  int Dd, E, dtr;
};

static void run_mamba_stack(float* x, int Bn, int L, const MParams& P,
                            float* xz, float* xc, float* xdbl, float* dtb, float* yb,
                            hipStream_t stream)
{
  const int rows = Bn * L;
  const int E = P.E, Dd = P.Dd, dtr = P.dtr;
  const int twoE = 2 * E;
  const int ldx  = dtr + 32;   // dt_rank + 2*d_state
  for (int i = 0; i < 2; ++i) {
    const float* inw    = P.in_w   + (size_t)i * twoE * Dd;
    const float* cw     = P.conv_w + (size_t)i * E * 4;
    const float* cb     = P.conv_b + (size_t)i * E;
    const float* xpw    = P.xproj_w+ (size_t)i * ldx * E;
    const float* dtw    = P.dt_w   + (size_t)i * E * dtr;
    const float* dtbias = P.dt_b   + (size_t)i * E;
    const float* alog   = P.Alog   + (size_t)i * E * 16;
    const float* Dp     = P.D      + (size_t)i * E;
    const float* outw   = P.out_w  + (size_t)i * Dd * E;

    // xz = x @ in_w^T                                (rows x 2E)
    gemm_nt_wmma<<<dim3(cdiv(twoE, 64), cdiv(rows, 16)), 32, 0, stream>>>(
        x, Dd, inw, Dd, xz, twoE, rows, twoE, Dd, nullptr, nullptr, 0);
    // depthwise conv + SiLU                          (rows x E)
    int tE = rows * E;
    dwconv_silu_k<<<cdiv(tE, 256), 256, 0, stream>>>(xz, twoE, E, L, cw, cb, xc, tE);
    // x_dbl = xc @ xproj_w^T                         (rows x ldx)
    gemm_nt_wmma<<<dim3(cdiv(ldx, 64), cdiv(rows, 16)), 32, 0, stream>>>(
        xc, E, xpw, E, xdbl, ldx, rows, ldx, E, nullptr, nullptr, 0);
    // dt = softplus(x_dbl[:, :dtr] @ dt_w^T + dt_b)  (rows x E)
    gemm_nt_wmma<<<dim3(cdiv(E, 64), cdiv(rows, 16)), 32, 0, stream>>>(
        xdbl, ldx, dtw, dtr, dtb, E, rows, E, dtr, dtbias, nullptr, 1);
    // selective scan -> y                            (rows x E)
    int tBE = Bn * E;
    sscan_k<<<cdiv(tBE, 256), 256, 0, stream>>>(dtb, xdbl, ldx, dtr, xc, alog, Dp,
                                                yb, Bn, L, E);
    // y *= silu(z)
    gate_silu_k<<<cdiv(tE, 256), 256, 0, stream>>>(yb, xz, twoE, E, tE);
    // x = y @ out_w^T + x   (residual fused)
    gemm_nt_wmma<<<dim3(cdiv(Dd, 64), cdiv(rows, 16)), 32, 0, stream>>>(
        yb, E, outw, E, x, Dd, rows, Dd, E, nullptr, x, 0);
  }
}

// ---------------------------------------------------------------------------
// Entry point
// ---------------------------------------------------------------------------
extern "C" void kernel_launch(void* const* d_in, const int* in_sizes, int n_in,
                              void* d_out, int out_size, void* d_ws, size_t ws_size,
                              hipStream_t stream)
{
  const float* x_cwt        = (const float*)d_in[0];
  const float* x_features   = (const float*)d_in[1];
  const float* patch_conv_w = (const float*)d_in[2];
  const float* patch_conv_b = (const float*)d_in[3];
  const float* patch_proj_w = (const float*)d_in[4];
  const float* patch_proj_b = (const float*)d_in[5];
  const float* proj_w       = (const float*)d_in[6];
  const float* proj_b       = (const float*)d_in[7];
  const float* focus        = (const float*)d_in[8];
  const float* ln_w         = (const float*)d_in[9];
  const float* ln_b         = (const float*)d_in[10];
  MParams m1 = {(const float*)d_in[11], (const float*)d_in[12], (const float*)d_in[13],
                (const float*)d_in[14], (const float*)d_in[15], (const float*)d_in[16],
                (const float*)d_in[17], (const float*)d_in[18], (const float*)d_in[19],
                768, 1536, 48};
  MParams m2 = {(const float*)d_in[20], (const float*)d_in[21], (const float*)d_in[22],
                (const float*)d_in[23], (const float*)d_in[24], (const float*)d_in[25],
                (const float*)d_in[26], (const float*)d_in[27], (const float*)d_in[28],
                128, 256, 8};
  const float* cls_w1 = (const float*)d_in[29];
  const float* cls_b1 = (const float*)d_in[30];
  const float* cls_w2 = (const float*)d_in[31];
  const float* cls_b2 = (const float*)d_in[32];

  float* ws = (float*)d_ws;
  size_t off = 0;
  auto alloc = [&](size_t n) { float* p = ws + off; off += n; return p; };

  const size_t CCN = 8ull * 128 * 768;          // 786432 (also == 8*768*128)
  float* xp3  = alloc(1024ull * 196);
  float* xp2  = alloc(1024ull * 256);
  float* xpr  = alloc(1024ull * 256);
  float* cc   = alloc(CCN);
  float* x1a  = alloc(CCN);                     // accumulator / forward-stack buffer
  float* bufB = alloc(CCN);
  float* bufC = alloc(CCN);
  float* xz   = alloc(1024ull * 3072);          // == 6144*512
  float* xcb  = alloc(1024ull * 1536);          // == 6144*256
  float* xdbl = alloc(6144ull * 40);            // max(1024*80, 6144*40)
  float* dtb  = alloc(1024ull * 1536);
  float* yb   = alloc(1024ull * 1536);
  float* x3   = alloc(8ull * 768);
  float* h1   = alloc(8ull * 384);

  // 1) patchify conv (implicit WMMA GEMM) -> xp3 (8,128,196)
  patchify_wmma<<<dim3(13, 8, 8), 32, 0, stream>>>(x_cwt, patch_conv_w,
                                                   patch_conv_b, xp3);
  // 2) xp2 = xp3 @ patch_proj_w^T + b     (1024 x 256, K=196)
  gemm_nt_wmma<<<dim3(4, 64), 32, 0, stream>>>(xp3, 196, patch_proj_w, 196,
                                               xp2, 256, 1024, 256, 196,
                                               patch_proj_b, nullptr, 0);
  // 3) xproj = x_features @ proj_w^T + b  (1024 x 256, K=512)
  gemm_nt_wmma<<<dim3(4, 64), 32, 0, stream>>>(x_features, 512, proj_w, 512,
                                               xpr, 256, 1024, 256, 512,
                                               proj_b, nullptr, 0);
  // 4) focus-mix + gelu + concat + layernorm -> cc (8,128,768)
  build_cc_ln_k<<<1024, 256, 0, stream>>>(xp2, xpr, focus, ln_w, ln_b, cc);

  const int tCC = (int)CCN;
  const int t4  = tCC / 4;

  // 5) mamba-1 forward on x1a  (copy staged through LDS with async loads)
  async_copy_k<<<cdiv(t4, 256), 256, 0, stream>>>((const float4*)cc,
                                                  (float4*)x1a, t4);
  run_mamba_stack(x1a, 8, 128, m1, xz, xcb, xdbl, dtb, yb, stream);
  // 6) mamba-1 on flipped seq, then x1 += flip(result)
  flip_copy_k<<<cdiv(tCC, 256), 256, 0, stream>>>(cc, bufB, 128, 768, tCC);
  run_mamba_stack(bufB, 8, 128, m1, xz, xcb, xdbl, dtb, yb, stream);
  flip_add_k<<<cdiv(tCC, 256), 256, 0, stream>>>(x1a, bufB, 128, 768, tCC);

  // 7) ccp = transpose(cc) -> (8,768,128); mamba-2 forward
  transpose12_k<<<cdiv(tCC, 256), 256, 0, stream>>>(cc, bufC, 128, 768, tCC);
  tdm_copy_k<<<cdiv(tCC, TDM_TILE), 32, 0, stream>>>(bufC, bufB, tCC);
  run_mamba_stack(bufB, 8, 768, m2, xz, xcb, xdbl, dtb, yb, stream);
  // 8) mamba-2 on seq-flipped ccp (reuse cc as buffer)
  flip_copy_k<<<cdiv(tCC, 256), 256, 0, stream>>>(bufC, cc, 768, 128, tCC);
  run_mamba_stack(cc, 8, 768, m2, xz, xcb, xdbl, dtb, yb, stream);
  // 9) x += transpose(x2_fwd) + flip_axis1(transpose(x2_rev))
  add_trans_k<<<cdiv(tCC, 256), 256, 0, stream>>>(x1a, bufB, 128, 768, 0, tCC);
  add_trans_k<<<cdiv(tCC, 256), 256, 0, stream>>>(x1a, cc,   128, 768, 1, tCC);

  // 10) mean over axis 1 -> x3 (8,768)
  mean_axis1_k<<<cdiv(8 * 768, 256), 256, 0, stream>>>(x1a, x3, 8, 128, 768);
  // 11) classifier head
  gemm_nt_wmma<<<dim3(6, 1), 32, 0, stream>>>(x3, 768, cls_w1, 768, h1, 384,
                                              8, 384, 768, cls_b1, nullptr, 0);
  gemm_nt_wmma<<<dim3(1, 1), 32, 0, stream>>>(h1, 384, cls_w2, 384,
                                              (float*)d_out, 10,
                                              8, 10, 384, cls_b2, nullptr, 0);
}
